// MyTransformerLayer_90452011254120
// MI455X (gfx1250) — matrix operand order repaired and checked by reading, compiled, not run on hardware
//
#include <hip/hip_runtime.h>
#include <hip/hip_bf16.h>
#include <stdint.h>
#include <math.h>

// MyTransformerLayer for MI455X (gfx1250): bf16 WMMA GEMMs + flash attention,
// TDM (tensor_load_to_lds) for weight tiles. B=4,N=2048,E=256,H=4,DH=64.

#define B_ 4
#define N_ 2048
#define E_ 256
#define H_ 4
#define DH_ 64
#define LN_EPS 1e-5f

typedef __attribute__((ext_vector_type(16))) __bf16 bf16x16;
typedef __attribute__((ext_vector_type(8))) float f32x8;
typedef __attribute__((ext_vector_type(4))) uint32_t u32x4;
typedef __attribute__((ext_vector_type(8))) int i32x8;
typedef __attribute__((ext_vector_type(4))) int i32x4;

union FragU { uint32_t u[8]; bf16x16 v; };

// native hardware f32->bf16 convert (RNE on gfx1250)
__device__ __forceinline__ uint16_t f2bf(float f) {
  union { __bf16 b; uint16_t u; } c;
  c.b = (__bf16)f;
  return c.u;
}
__device__ __forceinline__ uint32_t pack2(float lo, float hi) {
  union { __bf16 b[2]; uint32_t u; } c;
  c.b[0] = (__bf16)lo;
  c.b[1] = (__bf16)hi;
  return c.u;
}

// f32 [K,Nc] -> bf16 transposed [Nc,K]
__global__ void cvt_transpose_bf16(const float* __restrict__ in,
                                   uint16_t* __restrict__ out, int K, int Nc) {
  int i = blockIdx.x * blockDim.x + threadIdx.x;
  if (i >= K * Nc) return;
  int k = i / Nc, n = i - k * Nc;
  out[(size_t)n * K + k] = f2bf(in[i]);
}

// ---------------------------------------------------------------------------
// GEMM: Y[M,Nc] = X[M,K] @ W[K,Nc] + bias (+res). W given as bf16 WT[Nc,K].
// Block tile 128x128, K-step 32, 8 waves each computing 32(M) x 64(N).
// B tile DMA'd by the Tensor Data Mover (with LDS row padding done by TDM).
// ---------------------------------------------------------------------------
#define GBM 128
#define GBN 128
#define GBK 32
#define GPAD 8

__global__ __launch_bounds__(256) void gemm_bf16_wmma(
    const float* __restrict__ X, const uint16_t* __restrict__ WT,
    const float* __restrict__ bias, const float* __restrict__ res,
    float* __restrict__ Y, int M, int K, int Nc) {
  __shared__ uint16_t sA[GBM][GBK + GPAD];   // [m][k], row stride 80B
  __shared__ uint16_t sB[GBN][GBK + GPAD];   // [n][k], row stride 80B
  const int tid = threadIdx.x;
  const int lane = tid & 31, wid = tid >> 5;
  const int lrow = lane & 15, lhi = lane >> 4;
  const int waveM = wid >> 1, waveN = wid & 1;
  const int m0 = blockIdx.y * GBM, n0 = blockIdx.x * GBN;

  const f32x8 vz = {0.f, 0.f, 0.f, 0.f, 0.f, 0.f, 0.f, 0.f};
  f32x8 acc[2][4];
#pragma unroll
  for (int i = 0; i < 2; i++)
#pragma unroll
    for (int j = 0; j < 4; j++) acc[i][j] = vz;

  const int arow = tid >> 1, acolb = (tid & 1) * 16;   // A: 128 rows x 32 cols
  const uint32_t ldsB = (uint32_t)(uintptr_t)(void*)&sB[0][0];

  for (int kt = 0; kt < K; kt += GBK) {
    if (tid < 32) {
      // D# group0: count=1, lds_addr, global tile addr, type=2
      uint64_t ga = (uint64_t)(uintptr_t)WT + 2ull * ((size_t)n0 * K + kt);
      u32x4 g0 = {1u, ldsB, (uint32_t)ga,
                  (uint32_t)((ga >> 32) & 0x1FFFFFFull) | (2u << 30)};
      // D# group1: data_size=2B, pad after 16 DWORDs by 4 DWORDs,
      // tensor_dim0=K, tensor_dim1=Nc, tile 32x128, dim0_stride=K
      uint32_t w0 = (1u << 16) | (1u << 20) | (3u << 22) | (3u << 25);
      uint32_t w1 = ((uint32_t)K & 0xFFFFu) << 16;
      uint32_t w2 = (((uint32_t)K >> 16) & 0xFFFFu) | (((uint32_t)Nc & 0xFFFFu) << 16);
      uint32_t w3 = (((uint32_t)Nc >> 16) & 0xFFFFu) | (32u << 16);
      uint32_t w4 = 128u;
      uint32_t w5 = (uint32_t)K;
      i32x8 g1 = {(int)w0, (int)w1, (int)w2, (int)w3, (int)w4, (int)w5, 0, 0};
      i32x4 z4 = {0, 0, 0, 0};
      i32x8 z8 = {0, 0, 0, 0, 0, 0, 0, 0};
      __builtin_amdgcn_tensor_load_to_lds(g0, g1, z4, z4, z8, 0);
    }
    // A tile: f32 -> bf16 convert (overlaps the B DMA)
    {
      const float4* gA = (const float4*)(X + (size_t)(m0 + arow) * K + kt + acolb);
#pragma unroll
      for (int j4 = 0; j4 < 4; j4++) {
        float4 t = gA[j4];
        *(uint32_t*)&sA[arow][acolb + 4 * j4 + 0] = pack2(t.x, t.y);
        *(uint32_t*)&sA[arow][acolb + 4 * j4 + 2] = pack2(t.z, t.w);
      }
    }
    if (tid < 32) __builtin_amdgcn_s_wait_tensorcnt(0);
    __syncthreads();

    FragU af[2];
#pragma unroll
    for (int i = 0; i < 2; i++) {
      const uint32_t* p = (const uint32_t*)&sA[waveM * 32 + i * 16 + lrow][0];
#pragma unroll
      for (int vg = 0; vg < 8; vg++) {
        int idx = lhi * 8 + vg;
        af[i].u[vg] = p[idx];
      }
    }
    FragU bf[4];
#pragma unroll
    for (int j = 0; j < 4; j++) {
      const uint32_t* p = (const uint32_t*)&sB[waveN * 64 + j * 16 + lrow][lhi * 16];
#pragma unroll
      for (int q = 0; q < 8; q++) bf[j].u[q] = p[q];
    }
#pragma unroll
    for (int i = 0; i < 2; i++)
#pragma unroll
      for (int j = 0; j < 4; j++)
        acc[i][j] = __builtin_amdgcn_wmma_f32_16x16x32_bf16(
            false, af[i].v, false, bf[j].v, (short)0, acc[i][j], false, false);
    __syncthreads();
  }

  // epilogue: batched loads, hoisted residual branch
  const int colb = n0 + waveN * 64 + lrow;
  float bv[4];
#pragma unroll
  for (int j = 0; j < 4; j++) bv[j] = bias[colb + j * 16];
#pragma unroll
  for (int i = 0; i < 2; i++) {
    const int rowb = m0 + waveM * 32 + i * 16 + lhi * 8;
    const size_t base = (size_t)rowb * Nc + colb;
    if (res) {
      float rv[4][8];
#pragma unroll
      for (int j = 0; j < 4; j++)
#pragma unroll
        for (int r = 0; r < 8; r++) rv[j][r] = res[base + (size_t)r * Nc + j * 16];
#pragma unroll
      for (int j = 0; j < 4; j++)
#pragma unroll
        for (int r = 0; r < 8; r++)
          Y[base + (size_t)r * Nc + j * 16] = acc[i][j][r] + bv[j] + rv[j][r];
    } else {
#pragma unroll
      for (int j = 0; j < 4; j++)
#pragma unroll
        for (int r = 0; r < 8; r++)
          Y[base + (size_t)r * Nc + j * 16] = acc[i][j][r] + bv[j];
    }
  }
}

// ---------------------------------------------------------------------------
// Flash attention. Q,K,V,O all [B, N, H*DH] (head-major columns). One block
// handles 128 query rows of one (b,h); 8 waves x 16 rows. Key tile = 64.
// ---------------------------------------------------------------------------
#define ABM 128
#define ABN 64

__global__ __launch_bounds__(256) void attn_wmma(
    const float* __restrict__ Q, const float* __restrict__ Km,
    const float* __restrict__ V, float* __restrict__ O, float scale) {
  __shared__ uint16_t sK[ABN][DH_ + 4];      // [key][d]
  __shared__ uint16_t sVt[DH_][ABN + 4];     // [d][key]
  __shared__ uint16_t sP[8][16][ABN + 4];    // wave-private P tiles
  const int tid = threadIdx.x, lane = tid & 31, wid = tid >> 5;
  const int lrow = lane & 15, lhi = lane >> 4;
  const int b = blockIdx.z, h = blockIdx.y;
  const int q0 = blockIdx.x * ABM + wid * 16;
  const size_t baseBH = ((size_t)b * N_) * E_ + (size_t)h * DH_;

  FragU qf[2];
  {
    const float* qb = Q + baseBH + (size_t)(q0 + lrow) * E_;
#pragma unroll
    for (int s = 0; s < 2; s++)
#pragma unroll
      for (int e = 0; e < 16; e += 2) {
        int k = s * 32 + lhi * 16 + e;
        qf[s].u[e >> 1] = pack2(qb[k] * scale, qb[k + 1] * scale);
      }
  }

  float mrun[8], lrun[8];
#pragma unroll
  for (int r = 0; r < 8; r++) { mrun[r] = -3.0e38f; lrun[r] = 0.f; }
  const f32x8 vz = {0.f, 0.f, 0.f, 0.f, 0.f, 0.f, 0.f, 0.f};
  f32x8 acc[4];
#pragma unroll
  for (int t = 0; t < 4; t++) acc[t] = vz;

  const int ldr = tid >> 2;          // key row 0..63
  const int ldc = (tid & 3) * 16;    // d col base

  for (int kt = 0; kt < N_; kt += ABN) {
    const float4* gK = (const float4*)(Km + baseBH + (size_t)(kt + ldr) * E_ + ldc);
    const float4* gV = (const float4*)(V + baseBH + (size_t)(kt + ldr) * E_ + ldc);
#pragma unroll
    for (int j4 = 0; j4 < 4; j4++) {
      float4 t = gK[j4];
      *(uint32_t*)&sK[ldr][ldc + 4 * j4 + 0] = pack2(t.x, t.y);
      *(uint32_t*)&sK[ldr][ldc + 4 * j4 + 2] = pack2(t.z, t.w);
    }
#pragma unroll
    for (int j4 = 0; j4 < 4; j4++) {
      float4 t = gV[j4];
      sVt[ldc + 4 * j4 + 0][ldr] = f2bf(t.x);
      sVt[ldc + 4 * j4 + 1][ldr] = f2bf(t.y);
      sVt[ldc + 4 * j4 + 2][ldr] = f2bf(t.z);
      sVt[ldc + 4 * j4 + 3][ldr] = f2bf(t.w);
    }
    __syncthreads();

    // S = (Q*scale) @ K^T : 4 key-tiles of 16
    f32x8 sc[4];
#pragma unroll
    for (int j = 0; j < 4; j++) {
      sc[j] = vz;
#pragma unroll
      for (int s = 0; s < 2; s++) {
        FragU kf;
        const uint32_t* p = (const uint32_t*)&sK[j * 16 + lrow][s * 32 + lhi * 16];
#pragma unroll
        for (int q = 0; q < 8; q++) kf.u[q] = p[q];
        sc[j] = __builtin_amdgcn_wmma_f32_16x16x32_bf16(
            false, qf[s].v, false, kf.v, (short)0, sc[j], false, false);
      }
    }

    // online softmax (rows = lhi*8+r; 16 columns spread across a lane half)
#pragma unroll
    for (int r = 0; r < 8; r++) {
      float pm = fmaxf(fmaxf(sc[0][r], sc[1][r]), fmaxf(sc[2][r], sc[3][r]));
#pragma unroll
      for (int m = 1; m < 16; m <<= 1) pm = fmaxf(pm, __shfl_xor(pm, m, 32));
      float mnew = fmaxf(mrun[r], pm);
      float corr = __expf(mrun[r] - mnew);
      mrun[r] = mnew;
      lrun[r] *= corr;
#pragma unroll
      for (int t = 0; t < 4; t++) acc[t][r] *= corr;
      float psum = 0.f;
#pragma unroll
      for (int j = 0; j < 4; j++) {
        float p = __expf(sc[j][r] - mnew);
        psum += p;
        sP[wid][lhi * 8 + r][j * 16 + lrow] = f2bf(p);
      }
#pragma unroll
      for (int m = 1; m < 16; m <<= 1) psum += __shfl_xor(psum, m, 32);
      lrun[r] += psum;
    }

    // acc += P @ V
#pragma unroll
    for (int s2 = 0; s2 < 2; s2++) {
      FragU pf;
      const uint32_t* pp = (const uint32_t*)&sP[wid][lrow][0];
#pragma unroll
      for (int vg = 0; vg < 8; vg++) {
        int idx = s2 * 16 + lhi * 8 + vg;
        pf.u[vg] = pp[idx];
      }
#pragma unroll
      for (int t = 0; t < 4; t++) {
        FragU vf;
        const uint32_t* pv = (const uint32_t*)&sVt[t * 16 + lrow][s2 * 32 + lhi * 16];
#pragma unroll
        for (int q = 0; q < 8; q++) vf.u[q] = pv[q];
        acc[t] = __builtin_amdgcn_wmma_f32_16x16x32_bf16(
            false, pf.v, false, vf.v, (short)0, acc[t], false, false);
      }
    }
    __syncthreads();
  }

#pragma unroll
  for (int r = 0; r < 8; r++) {
    float inv = 1.f / lrun[r];
    int row = q0 + lhi * 8 + r;
    float* ob = O + baseBH + (size_t)row * E_;
#pragma unroll
    for (int t = 0; t < 4; t++) ob[t * 16 + lrow] = acc[t][r] * inv;
  }
}

// ---------------------------------------------------------------------------
// qkv [B,N,3E] (inner layout (h,d,3)) -> rotary(q), rotary(k), v in [B,N,H*DH]
// ---------------------------------------------------------------------------
__global__ void rotary_split(const float* __restrict__ qkv,
                             const float* __restrict__ enc,
                             float* __restrict__ q, float* __restrict__ k,
                             float* __restrict__ v) {
  int i = blockIdx.x * blockDim.x + threadIdx.x;  // B*N*H*DH/2
  int j = i & 31;
  int h = (i >> 5) & (H_ - 1);
  int n = (i >> 7) & (N_ - 1);
  int b = i >> 18;
  size_t base = ((size_t)b * N_ + n) * (3 * E_) + (size_t)h * (DH_ * 3);
  int d0 = 2 * j;
  float q0v = qkv[base + (size_t)d0 * 3 + 0];
  float q1v = qkv[base + (size_t)(d0 + 1) * 3 + 0];
  float k0v = qkv[base + (size_t)d0 * 3 + 1];
  float k1v = qkv[base + (size_t)(d0 + 1) * 3 + 1];
  float v0v = qkv[base + (size_t)d0 * 3 + 2];
  float v1v = qkv[base + (size_t)(d0 + 1) * 3 + 2];
  size_t co = ((size_t)b * N_ + n) * DH_ + d0;
  size_t so = co + (size_t)B_ * N_ * DH_;
  float c0 = enc[co], c1 = enc[co + 1], s0 = enc[so], s1 = enc[so + 1];
  size_t ob = ((size_t)b * N_ + n) * E_ + (size_t)h * DH_ + d0;
  q[ob]     = q0v * c0 - q1v * s0;
  q[ob + 1] = q1v * c1 + q0v * s1;
  k[ob]     = k0v * c0 - k1v * s0;
  k[ob + 1] = k1v * c1 + k0v * s1;
  v[ob] = v0v;
  v[ob + 1] = v1v;
}

// LayerNorm (biased var) + exact GELU, in place over rows of 512
__global__ __launch_bounds__(256) void ln_gelu(float* __restrict__ hbuf,
                                               const float* __restrict__ g,
                                               const float* __restrict__ bn) {
  __shared__ float rs[256], rq[256];
  int row = blockIdx.x, t = threadIdx.x;
  float* hp = hbuf + (size_t)row * 512;
  float a0 = hp[t], a1 = hp[t + 256];
  rs[t] = a0 + a1;
  rq[t] = a0 * a0 + a1 * a1;
  __syncthreads();
  for (int off = 128; off > 0; off >>= 1) {
    if (t < off) { rs[t] += rs[t + off]; rq[t] += rq[t + off]; }
    __syncthreads();
  }
  float mean = rs[0] * (1.f / 512.f);
  float var = rq[0] * (1.f / 512.f) - mean * mean;
  float inv = rsqrtf(var + LN_EPS);
  float y0 = (a0 - mean) * inv * g[t] + bn[t];
  float y1 = (a1 - mean) * inv * g[t + 256] + bn[t + 256];
  hp[t]       = 0.5f * y0 * (1.f + erff(y0 * 0.70710678118654752f));
  hp[t + 256] = 0.5f * y1 * (1.f + erff(y1 * 0.70710678118654752f));
}

__global__ void concat2(const float* __restrict__ a, const float* __restrict__ bb,
                        float* __restrict__ out) {
  int i = blockIdx.x * blockDim.x + threadIdx.x;  // M*512
  int r = i >> 9, c = i & 511;
  out[i] = (c < 256) ? a[(size_t)r * 256 + c] : bb[(size_t)r * 256 + (c - 256)];
}

__global__ void finalize_out(const float* __restrict__ x0,
                             const float* __restrict__ x1,
                             float* __restrict__ out) {
  int i = blockIdx.x * blockDim.x + threadIdx.x;  // M*512
  int r = i >> 9, c = i & 511;
  float v = (c < 256) ? x0[(size_t)r * 256 + c] : x1[(size_t)r * 256 + (c - 256)];
  out[i] = v;
  out[(size_t)i + (size_t)B_ * N_ * 512] = v;
}

// ---------------------------------------------------------------------------
extern "C" void kernel_launch(void* const* d_in, const int* in_sizes, int n_in,
                              void* d_out, int out_size, void* d_ws, size_t ws_size,
                              hipStream_t stream) {
  (void)in_sizes; (void)n_in; (void)out_size; (void)ws_size;
  const float* desc0 = (const float*)d_in[0];
  const float* desc1 = (const float*)d_in[1];
  const float* enc0  = (const float*)d_in[2];
  const float* enc1  = (const float*)d_in[3];
  const float* sWqkv = (const float*)d_in[4];
  const float* sbqkv = (const float*)d_in[5];
  const float* sWout = (const float*)d_in[6];
  const float* sbout = (const float*)d_in[7];
  const float* sW1 = (const float*)d_in[8];
  const float* sb1 = (const float*)d_in[9];
  const float* sg  = (const float*)d_in[10];
  const float* sbn = (const float*)d_in[11];
  const float* sW2 = (const float*)d_in[12];
  const float* sb2 = (const float*)d_in[13];
  const float* cWqk = (const float*)d_in[14];
  const float* cbqk = (const float*)d_in[15];
  const float* cWv = (const float*)d_in[16];
  const float* cbv = (const float*)d_in[17];
  const float* cWo = (const float*)d_in[18];
  const float* cbo = (const float*)d_in[19];
  const float* cW1 = (const float*)d_in[20];
  const float* cb1 = (const float*)d_in[21];
  const float* cg  = (const float*)d_in[22];
  const float* cbn = (const float*)d_in[23];
  const float* cW2 = (const float*)d_in[24];
  const float* cb2 = (const float*)d_in[25];

  const size_t S1 = (size_t)B_ * N_ * E_;  // 2M floats
  float* w = (float*)d_ws;
  size_t o = 0;
  float* bD0  = w + o; o += S1;
  float* bD1  = w + o; o += S1;
  float* bQKV = w + o; o += 3 * S1;   // qkv (live early) / cat (live late)
  float* bCAT = bQKV;
  float* bH1  = w + o; o += 2 * S1;
  float* bQ   = w + o; o += S1;
  float* bK   = w + o; o += S1;
  float* bV   = w + o; o += S1;
  float* bV2  = w + o; o += S1;
  float* bCTX = w + o; o += S1;
  float* bCT2 = w + o; o += S1;
  float* bMSG = w + o; o += S1;

  // bf16 transposed weights [Nc][K]
  uint16_t* u = (uint16_t*)(w + o);
  size_t uo = 0;
  auto carve = [&](size_t n) { uint16_t* p = u + uo; uo += n; return p; };
  uint16_t* tWqkv = carve(768 * 256);
  uint16_t* tWout = carve(256 * 256);
  uint16_t* tW1   = carve(512 * 512);
  uint16_t* tW2   = carve(256 * 512);
  uint16_t* tcWqk = carve(256 * 256);
  uint16_t* tcWv  = carve(256 * 256);
  uint16_t* tcWo  = carve(256 * 256);
  uint16_t* tcW1  = carve(512 * 512);
  uint16_t* tcW2  = carve(256 * 512);

  auto cvt = [&](const float* in, uint16_t* outp, int K, int Nc) {
    int tot = K * Nc;
    cvt_transpose_bf16<<<(tot + 255) / 256, 256, 0, stream>>>(in, outp, K, Nc);
  };
  cvt(sWqkv, tWqkv, 256, 768);
  cvt(sWout, tWout, 256, 256);
  cvt(sW1, tW1, 512, 512);
  cvt(sW2, tW2, 512, 256);
  cvt(cWqk, tcWqk, 256, 256);
  cvt(cWv, tcWv, 256, 256);
  cvt(cWo, tcWo, 256, 256);
  cvt(cW1, tcW1, 512, 512);
  cvt(cW2, tcW2, 512, 256);

  const int M = B_ * N_;
  const float scale = 0.125f;  // 1/sqrt(DH); cross uses (DH^-.25)^2 == same
  dim3 attg(N_ / ABM, H_, B_);

  auto gemm = [&](const float* X, const uint16_t* WT, const float* bias,
                  const float* res, float* Y, int K, int Nc) {
    dim3 g(Nc / GBN, M / GBM);
    gemm_bf16_wmma<<<g, dim3(256), 0, stream>>>(X, WT, bias, res, Y, M, K, Nc);
  };

  // ---- self blocks (shared weights) ----
  const float* descs[2] = {desc0, desc1};
  const float* encs[2]  = {enc0, enc1};
  float* outs[2] = {bD0, bD1};
  for (int i = 0; i < 2; i++) {
    gemm(descs[i], tWqkv, sbqkv, nullptr, bQKV, E_, 3 * E_);
    rotary_split<<<(B_ * N_ * H_ * DH_ / 2) / 256, 256, 0, stream>>>(bQKV, encs[i], bQ, bK, bV);
    attn_wmma<<<attg, 256, 0, stream>>>(bQ, bK, bV, bCTX, scale);
    gemm(bCTX, tWout, sbout, nullptr, bMSG, E_, E_);
    concat2<<<(M * 512) / 256, 256, 0, stream>>>(descs[i], bMSG, bCAT);
    gemm(bCAT, tW1, sb1, nullptr, bH1, 2 * E_, 2 * E_);
    ln_gelu<<<M, 256, 0, stream>>>(bH1, sg, sbn);
    gemm(bH1, tW2, sb2, descs[i], outs[i], 2 * E_, E_);
  }

  // ---- cross block ----
  gemm(bD0, tcWqk, cbqk, nullptr, bQ, E_, E_);    // qk0
  gemm(bD1, tcWqk, cbqk, nullptr, bK, E_, E_);    // qk1
  gemm(bD0, tcWv, cbv, nullptr, bV, E_, E_);      // v0
  gemm(bD1, tcWv, cbv, nullptr, bV2, E_, E_);     // v1
  attn_wmma<<<attg, 256, 0, stream>>>(bQ, bK, bV2, bCTX, scale);  // m0 = A01 @ v1
  attn_wmma<<<attg, 256, 0, stream>>>(bK, bQ, bV, bCT2, scale);   // m1 = A10 @ v0

  // x0 = d0 + ffn(cat(d0, m0 @ Wo + bo))
  gemm(bCTX, tcWo, cbo, nullptr, bMSG, E_, E_);
  concat2<<<(M * 512) / 256, 256, 0, stream>>>(bD0, bMSG, bCAT);
  gemm(bCAT, tcW1, cb1, nullptr, bH1, 2 * E_, 2 * E_);
  ln_gelu<<<M, 256, 0, stream>>>(bH1, cg, cbn);
  gemm(bH1, tcW2, cb2, bD0, bQ, 2 * E_, E_);      // fin0 -> bQ (qk0 dead)

  // x1 = d1 + ffn(cat(d1, m1 @ Wo + bo))
  gemm(bCT2, tcWo, cbo, nullptr, bMSG, E_, E_);
  concat2<<<(M * 512) / 256, 256, 0, stream>>>(bD1, bMSG, bCAT);
  gemm(bCAT, tcW1, cb1, nullptr, bH1, 2 * E_, 2 * E_);
  ln_gelu<<<M, 256, 0, stream>>>(bH1, cg, cbn);
  gemm(bH1, tcW2, cb2, bD1, bK, 2 * E_, E_);      // fin1 -> bK (qk1 dead)

  finalize_out<<<(M * 512) / 256, 256, 0, stream>>>(bQ, bK, (float*)d_out);
}
